// HOIModel_old_32856499814726
// MI455X (gfx1250) — matrix operand order repaired and checked by reading, not compiled, not run on hardware
//
#include <hip/hip_runtime.h>

typedef __attribute__((ext_vector_type(16))) _Float16 v16h;
typedef __attribute__((ext_vector_type(4)))  _Float16 v4h;
typedef __attribute__((ext_vector_type(8)))  float    v8f;

#define PERSON_ID 1
#define KSEL 10
#define NSTRIPS 8

// ---------------- Kernel 1: per-image top-K selection (tiny) ----------------
__global__ void topk_kernel(const float* __restrict__ scores,
                            const int*   __restrict__ labels,
                            int* __restrict__ selidx,   // [8][20] (10 human, 10 object)
                            int* __restrict__ selok)    // [8][20]
{
    int b = threadIdx.x;
    if (b >= 8) return;
    bool used[100];
#pragma clang loop unroll(disable)
    for (int n = 0; n < 100; ++n) used[n] = false;
#pragma clang loop unroll(disable)
    for (int cat = 0; cat < 2; ++cat) {            // 0 = human, 1 = object
#pragma clang loop unroll(disable)
        for (int kk = 0; kk < KSEL; ++kk) {
            int best = -1; float bestv = -1e38f;
#pragma clang loop unroll(disable)
            for (int n = 0; n < 100; ++n) {
                if (used[n]) continue;
                bool isHuman = (labels[b*100 + n] == PERSON_ID);
                if ((cat == 0) != isHuman) continue;
                float sc = scores[b*100 + n];
                if (sc > bestv) { bestv = sc; best = n; }   // strict > == lowest-index tiebreak
            }
            int slot = b*20 + cat*KSEL + kk;
            if (best >= 0) { used[best] = true; selidx[slot] = best; selok[slot] = 1; }
            else           { selidx[slot] = 0;  selok[slot] = 0; }
        }
    }
}

// ---------------- Kernel 2a: partial box sums, b128 loads, 8 y-strips ----------------
// grid: 160 boxes * 8 strips; block = 192 threads (thread t owns channels 4t..4t+3)
__global__ void pool_partial_kernel(const float* __restrict__ features, // [8][96][96][768]
                                    const int*   __restrict__ boxes,    // [8][100][4]
                                    const int*   __restrict__ selidx,
                                    const int*   __restrict__ selok,
                                    float* __restrict__ partial)        // [160][8][768]
{
    int blk   = blockIdx.x;
    int strip = blk & (NSTRIPS - 1);
    int s     = blk >> 3;              // 0..159 == b*20 + slot20
    int b     = s / 20;
    int t     = threadIdx.x;           // 0..191

    float ax = 0.f, ay = 0.f, az = 0.f, aw = 0.f;
    if (selok[s]) {
        const int* bx = boxes + (size_t)(b*100 + selidx[s])*4;
        int x1 = bx[0], y1 = bx[1], x2 = bx[2], y2 = bx[3];
        for (int y = y1 + strip; y < y2; y += NSTRIPS) {
            const float* rowp = features + (size_t)((b*96 + y)*96)*768 + 4*t;
            int x = x1;
            // two independent b128 loads in flight per iteration
            for (; x + 1 < x2; x += 2) {
                float4 v0 = *(const float4*)(rowp + (size_t)x*768);
                float4 v1 = *(const float4*)(rowp + (size_t)(x+1)*768);
                ax += v0.x + v1.x; ay += v0.y + v1.y;
                az += v0.z + v1.z; aw += v0.w + v1.w;
            }
            if (x < x2) {
                float4 v = *(const float4*)(rowp + (size_t)x*768);
                ax += v.x; ay += v.y; az += v.z; aw += v.w;
            }
        }
    }
    float4* out = (float4*)(partial + ((size_t)s*NSTRIPS + strip)*768) + t;
    *out = make_float4(ax, ay, az, aw);
}

// ---------------- Kernel 2b: reduce strips, divide by area, convert f16 ----------------
// grid: 160 boxes; block = 192 threads
__global__ void pool_reduce_kernel(const float* __restrict__ partial,   // [160][8][768]
                                   const int*   __restrict__ boxes,
                                   const int*   __restrict__ selidx,
                                   const int*   __restrict__ selok,
                                   _Float16* __restrict__ pooled16)    // [160][768]
{
    int s = blockIdx.x;                // 0..159
    int b = s / 20;
    int t = threadIdx.x;               // 0..191
    float ax = 0.f, ay = 0.f, az = 0.f, aw = 0.f;
#pragma unroll
    for (int strip = 0; strip < NSTRIPS; ++strip) {
        float4 v = *(const float4*)(partial + ((size_t)s*NSTRIPS + strip)*768 + 4*t);
        ax += v.x; ay += v.y; az += v.z; aw += v.w;
    }
    float inv = 0.0f;
    if (selok[s]) {
        const int* bx = boxes + (size_t)(b*100 + selidx[s])*4;
        inv = 1.0f / (float)((bx[3] - bx[1]) * (bx[2] - bx[0]));
    }
    v4h o;
    o[0] = (_Float16)(ax * inv);
    o[1] = (_Float16)(ay * inv);
    o[2] = (_Float16)(az * inv);
    o[3] = (_Float16)(aw * inv);
    *(v4h*)(pooled16 + (size_t)s*768 + 4*t) = o;
}

// ---------------- Kernel 3: build pair rows X16[800][1536] + validity ----------------
// block = 192: each thread copies one uint4 (8 halves); 96 from human, 96 from object
__global__ void build_pairs_kernel(const _Float16* __restrict__ pooled16,
                                   const int* __restrict__ selok,
                                   _Float16* __restrict__ X16,      // [800][1536]
                                   int*      __restrict__ validrow, // [800]
                                   float*    __restrict__ valid_out)// d_out tail [800]
{
    int r = blockIdx.x;              // 0..799 : r = b*100 + i*10 + j
    int b = r / 100;
    int i = (r % 100) / 10;
    int j = r % 10;
    const uint4* hf = (const uint4*)(pooled16 + (size_t)(b*20 + i)*768);
    const uint4* of = (const uint4*)(pooled16 + (size_t)(b*20 + 10 + j)*768);
    uint4* xr = (uint4*)(X16 + (size_t)r*1536);
    int t = threadIdx.x;             // 0..191
    xr[t] = (t < 96) ? hf[t] : of[t - 96];
    if (t == 0) {
        int v = selok[b*20 + i] & selok[b*20 + 10 + j];
        validrow[r]  = v;
        valid_out[r] = v ? 1.0f : 0.0f;
    }
}

// ---------------- Kernel 4: f32 [K][N] -> f16 transposed [Nrows][K] (zero-padded) ----------------
__global__ void transpose_f32_to_f16(const float* __restrict__ W,
                                     _Float16* __restrict__ WT,
                                     int K, int N, int Nrows)
{
    int tid = blockIdx.x * blockDim.x + threadIdx.x;
    if (tid >= Nrows * K) return;
    int n = tid / K;
    int k = tid % K;
    float v = (n < N) ? W[(size_t)k*N + n] : 0.0f;
    WT[tid] = (_Float16)v;
}

// ---------------- Kernel 5: WMMA GEMM  C = act(A[M][K] @ BT[N][K]^T + bias) ----------------
// One 16x16 output tile per wave; 4 waves (128 threads) per block.
template<bool FINAL>
__global__ void wmma_gemm_kernel(const _Float16* __restrict__ A,   // [M][K] f16
                                 const _Float16* __restrict__ BT,  // [N][K] f16 (weights^T)
                                 const float*    __restrict__ bias,
                                 int M, int N, int K,
                                 _Float16* __restrict__ outH,      // !FINAL: [M][N] f16 (relu)
                                 float*    __restrict__ outF,      // FINAL:  [M][117] f32 masked
                                 const int* __restrict__ validrow,
                                 int NstoreCols)
{
    int wave = threadIdx.x >> 5;
    int lane = threadIdx.x & 31;
    int tile = blockIdx.x * (blockDim.x >> 5) + wave;
    int ntiles = N >> 4;
    if (tile >= (M >> 4) * ntiles) return;
    int mt = tile / ntiles;
    int nt = tile % ntiles;
    int m  = lane & 15;
    int hi = lane >> 4;

    const _Float16* Arow = A  + (size_t)(mt*16 + m) * K;  // A row m of tile
    const _Float16* Brow = BT + (size_t)(nt*16 + m) * K;  // column (nt*16+m) of B

    v8f acc = {};
    for (int k = 0; k < K; k += 32) {
        v16h a, bf;
        // A 16x32 f16 layout: halves 0-7 = K[k+hi*8 .. +7], halves 8-15 = K[k+16+hi*8 .. +7]
        ((uint4*)&a)[0]  = *(const uint4*)(Arow + k + hi*8);
        ((uint4*)&a)[1]  = *(const uint4*)(Arow + k + 16 + hi*8);
        // B 32x16 f16 layout: halves 0-7 = K[k+hi*8 .. +7], halves 8-15 = K[k+16+hi*8 .. +7] of column n
        ((uint4*)&bf)[0] = *(const uint4*)(Brow + k + hi*8);
        ((uint4*)&bf)[1] = *(const uint4*)(Brow + k + 16 + hi*8);
        acc = __builtin_amdgcn_wmma_f32_16x16x32_f16(false, a, false, bf,
                                                     (short)0, acc, false, false);
    }

    int col = nt*16 + m;                  // D: col = lane&15, row = r + 8*hi
    float bv = FINAL ? ((col < NstoreCols) ? bias[col] : 0.0f) : bias[col];
#pragma unroll
    for (int r = 0; r < 8; ++r) {
        int orow = mt*16 + r + 8*hi;
        float v = acc[r] + bv;
        if (!FINAL) {
            v = v > 0.0f ? v : 0.0f;
            outH[(size_t)orow*N + col] = (_Float16)v;
        } else if (col < NstoreCols) {
            outF[(size_t)orow*NstoreCols + col] = validrow[orow] ? v : 0.0f;
        }
    }
}

extern "C" void kernel_launch(void* const* d_in, const int* in_sizes, int n_in,
                              void* d_out, int out_size, void* d_ws, size_t ws_size,
                              hipStream_t stream)
{
    const float* features = (const float*)d_in[0]; // [8,96,96,768]
    const int*   boxes    = (const int*)  d_in[1]; // [8,100,4]
    const int*   labels   = (const int*)  d_in[2]; // [8,100]
    const float* scores   = (const float*)d_in[3]; // [8,100]
    const float* W1 = (const float*)d_in[4];       // [1536,512]
    const float* b1 = (const float*)d_in[5];
    const float* W2 = (const float*)d_in[6];       // [512,256]
    const float* b2 = (const float*)d_in[7];
    const float* W3 = (const float*)d_in[8];       // [256,117]
    const float* b3 = (const float*)d_in[9];

    float* rel_out   = (float*)d_out;              // [800][117]
    float* valid_out = rel_out + 800*117;          // [800]

    char* ws = (char*)d_ws;
    size_t off = 0;
    auto alloc = [&](size_t bytes) -> void* {
        void* p = ws + off;
        off = (off + bytes + 255) & ~(size_t)255;
        return p;
    };
    int*      selidx   = (int*)     alloc(160*4);
    int*      selok    = (int*)     alloc(160*4);
    float*    partial  = (float*)   alloc((size_t)160*NSTRIPS*768*4);
    _Float16* pooled16 = (_Float16*)alloc((size_t)160*768*2);
    _Float16* X16      = (_Float16*)alloc((size_t)800*1536*2);
    int*      validrow = (int*)     alloc(800*4);
    _Float16* W1T      = (_Float16*)alloc((size_t)512*1536*2);
    _Float16* W2T      = (_Float16*)alloc((size_t)256*512*2);
    _Float16* W3T      = (_Float16*)alloc((size_t)128*256*2);
    _Float16* H1       = (_Float16*)alloc((size_t)800*512*2);
    _Float16* H2       = (_Float16*)alloc((size_t)800*256*2);

    topk_kernel<<<1, 32, 0, stream>>>(scores, labels, selidx, selok);
    pool_partial_kernel<<<160*NSTRIPS, 192, 0, stream>>>(features, boxes, selidx, selok, partial);
    pool_reduce_kernel<<<160, 192, 0, stream>>>(partial, boxes, selidx, selok, pooled16);
    build_pairs_kernel<<<800, 192, 0, stream>>>(pooled16, selok, X16, validrow, valid_out);

    { int t = 512*1536; transpose_f32_to_f16<<<(t+255)/256, 256, 0, stream>>>(W1, W1T, 1536, 512, 512); }
    { int t = 256*512;  transpose_f32_to_f16<<<(t+255)/256, 256, 0, stream>>>(W2, W2T,  512, 256, 256); }
    { int t = 128*256;  transpose_f32_to_f16<<<(t+255)/256, 256, 0, stream>>>(W3, W3T,  256, 117, 128); }

    // Layer 1: [800,1536] x [1536,512] -> relu f16   (1600 tiles / 4 waves per block)
    wmma_gemm_kernel<false><<<400, 128, 0, stream>>>(X16, W1T, b1, 800, 512, 1536,
                                                     H1, nullptr, nullptr, 0);
    // Layer 2: [800,512] x [512,256] -> relu f16     (800 tiles)
    wmma_gemm_kernel<false><<<200, 128, 0, stream>>>(H1, W2T, b2, 800, 256, 512,
                                                     H2, nullptr, nullptr, 0);
    // Layer 3: [800,256] x [256,117(pad128)] -> masked f32  (400 tiles)
    wmma_gemm_kernel<true><<<100, 128, 0, stream>>>(H2, W3T, b3, 800, 128, 256,
                                                    nullptr, rel_out, validrow, 117);
}
